// CausalSelfAttention_10445360464224
// MI455X (gfx1250) — matrix operand order repaired and checked by reading, not compiled, not run on hardware
//
#include <hip/hip_runtime.h>
#include <hip/hip_bf16.h>

typedef __attribute__((ext_vector_type(16))) _Float16 v16h;
typedef __attribute__((ext_vector_type(8)))  _Float16 v8h;
typedef __attribute__((ext_vector_type(4)))  _Float16 v4h;
typedef __attribute__((ext_vector_type(8)))  float    v8f;

#define WMMA_F16(A, B, Cacc) \
    __builtin_amdgcn_wmma_f32_16x16x32_f16(false, (A), false, (B), (short)0, (Cacc), false, false)

#define CAT16(lo, hi) __builtin_shufflevector((lo), (hi), 0,1,2,3,4,5,6,7,8,9,10,11,12,13,14,15)

// max across each group of 16 lanes, pure-VALU DPP butterfly (no LDS traffic)
__device__ __forceinline__ float row16_max(float v) {
    int x;
    x = __builtin_amdgcn_update_dpp(0, __float_as_int(v), 0xB1, 0xf, 0xf, true);  // quad xor1
    v = fmaxf(v, __int_as_float(x));
    x = __builtin_amdgcn_update_dpp(0, __float_as_int(v), 0x4E, 0xf, 0xf, true);  // quad xor2
    v = fmaxf(v, __int_as_float(x));
    x = __builtin_amdgcn_update_dpp(0, __float_as_int(v), 0x141, 0xf, 0xf, true); // half mirror
    v = fmaxf(v, __int_as_float(x));
    x = __builtin_amdgcn_update_dpp(0, __float_as_int(v), 0x140, 0xf, 0xf, true); // row mirror
    v = fmaxf(v, __int_as_float(x));
    return v;
}

// ---------------------------------------------------------------------------
// fp32 -> f16 cast
// ---------------------------------------------------------------------------
__global__ void cast_f32_f16(const float* __restrict__ in, _Float16* __restrict__ out, int n4) {
    int i = blockIdx.x * blockDim.x + threadIdx.x;
    if (i >= n4) return;
    float4 v = ((const float4*)in)[i];
    v4h o;
    o[0] = (_Float16)v.x; o[1] = (_Float16)v.y; o[2] = (_Float16)v.z; o[3] = (_Float16)v.w;
    ((v4h*)out)[i] = o;
}

// ---------------------------------------------------------------------------
// out[m,n] = sum_k A[m,k] * W[n,k]  (both row-major along K).
// One wave: 32(M) x 64(N) tile = 8 WMMAs per 32-k chunk, ping-pong double
// buffering. 128 threads = 4 waves per block.
// TRANS_OUT stores out[(b*N+n)*Tdim + t] (for V).
// ---------------------------------------------------------------------------
template <typename OutT, bool TRANS_OUT>
__global__ __launch_bounds__(128, 1) void gemm_awt(const _Float16* __restrict__ A,
                                                   const _Float16* __restrict__ W,
                                                   OutT* __restrict__ out,
                                                   int M, int N, int K, int Tdim) {
    const int lane = threadIdx.x & 31;
    const int wave = threadIdx.x >> 5;
    const int half = lane >> 4;
    const int l16  = lane & 15;

    const int ntiles64 = N >> 6;
    int tile = blockIdx.x * 4 + wave;
    int mt = (tile / ntiles64) << 5;
    int nt = (tile % ntiles64) << 6;
    if (mt >= M) return;

    const _Float16* arow0 = A + (size_t)(mt + l16) * K + half * 8;
    const _Float16* arow1 = A + (size_t)(mt + 16 + l16) * K + half * 8;
    const _Float16* wp[4];
#pragma unroll
    for (int j = 0; j < 4; ++j) wp[j] = W + (size_t)(nt + j * 16 + l16) * K + half * 8;

    v8f acc[2][4] = {};

    auto loadA = [&](v16h (&dst)[2], int k) {
        dst[0] = CAT16(*(const v8h*)(arow0 + k), *(const v8h*)(arow0 + k + 16));
        dst[1] = CAT16(*(const v8h*)(arow1 + k), *(const v8h*)(arow1 + k + 16));
    };
    auto loadB = [&](v16h (&dst)[4], int k) {
#pragma unroll
        for (int j = 0; j < 4; ++j) dst[j] = CAT16(*(const v8h*)(wp[j] + k), *(const v8h*)(wp[j] + k + 16));
    };
    auto mmas = [&](const v16h (&a)[2], const v16h (&b)[4]) {
#pragma unroll
        for (int i = 0; i < 2; ++i)
#pragma unroll
            for (int j = 0; j < 4; ++j) acc[i][j] = WMMA_F16(a[i], b[j], acc[i][j]);
    };

    v16h aA[2], bA[4], aB[2], bB[4];
    loadA(aA, 0); loadB(bA, 0);
    int k0 = 0;
    for (; k0 < K - 64; k0 += 64) {                  // K is a multiple of 64
        loadA(aB, k0 + 32); loadB(bB, k0 + 32);
        mmas(aA, bA);
        loadA(aA, k0 + 64); loadB(bA, k0 + 64);
        mmas(aB, bB);
    }
    loadA(aB, k0 + 32); loadB(bB, k0 + 32);
    mmas(aA, bA);
    mmas(aB, bB);

    if (TRANS_OUT) {
        int bb = mt / Tdim, tb = mt % Tdim;
#pragma unroll
        for (int i = 0; i < 2; ++i)
#pragma unroll
            for (int j = 0; j < 4; ++j)
#pragma unroll
                for (int r = 0; r < 8; ++r)
                    out[((size_t)bb * N + nt + j * 16 + l16) * Tdim + tb + i * 16 + half * 8 + r] =
                        (OutT)acc[i][j][r];
    } else {
#pragma unroll
        for (int i = 0; i < 2; ++i)
#pragma unroll
            for (int j = 0; j < 4; ++j)
#pragma unroll
                for (int r = 0; r < 8; ++r)
                    out[(size_t)(mt + i * 16 + half * 8 + r) * N + nt + j * 16 + l16] =
                        (OutT)acc[i][j][r];
    }
}

// ---------------------------------------------------------------------------
// Online-softmax + P*V update for ONE 16-row tile given its two S sub-tiles.
// l accumulated by WMMA against an all-ones B fragment (replicated per lane).
// ---------------------------------------------------------------------------
__device__ __forceinline__ void softmax_pv(
    v8f s0, v8f s1, const v16h (&vb)[4], const v16h ones,
    _Float16* __restrict__ sp, int half, int l16,
    v8f (&o)[4], v8f& lacc, float (&mrow)[8]) {
    float alpha[8];
#pragma unroll
    for (int r = 0; r < 8; ++r) {
        float mnew = fmaxf(mrow[r], row16_max(fmaxf(s0[r], s1[r])));
        alpha[r] = __expf(mrow[r] - mnew);
        mrow[r]  = mnew;
    }
#pragma unroll
    for (int r = 0; r < 8; ++r) {
        sp[(half * 8 + r) * 32 + l16]      = (_Float16)__expf(s0[r] - mrow[r]);
        sp[(half * 8 + r) * 32 + 16 + l16] = (_Float16)__expf(s1[r] - mrow[r]);
    }
#pragma unroll
    for (int j = 0; j < 4; ++j)
#pragma unroll
        for (int r = 0; r < 8; ++r) o[j][r] *= alpha[r];
#pragma unroll
    for (int r = 0; r < 8; ++r) lacc[r] *= alpha[r];

    const _Float16* prow = sp + l16 * 32;
    v16h pf = CAT16(*(const v8h*)(prow + half * 16), *(const v8h*)(prow + half * 16 + 8));
#pragma unroll
    for (int j = 0; j < 4; ++j) o[j] = WMMA_F16(pf, vb[j], o[j]);
    lacc = WMMA_F16(pf, ones, lacc);
}

// ---------------------------------------------------------------------------
// One full (unmasked) 32-key step for a 32-row Q tile (two 16-row sub-tiles
// sharing the same K/V fragments): 8 S-WMMAs + 8 PV-WMMAs + 2 l-WMMAs.
// ---------------------------------------------------------------------------
template <int T, int C>
__device__ __forceinline__ void attn_step_full(
    const _Float16*& kp0, const _Float16*& kp1, const _Float16*& vtp,
    const v16h (&qa)[2][2], const v16h ones,
    _Float16* __restrict__ sp0, _Float16* __restrict__ sp1,
    int half, int l16,
    v8f (&o)[2][4], v8f (&lacc)[2], float (&mrow)[2][8]) {
    v16h kb00 = *(const v16h*)(kp0);
    v16h kb01 = *(const v16h*)(kp0 + 32);
    v16h kb10 = *(const v16h*)(kp1);
    v16h kb11 = *(const v16h*)(kp1 + 32);
    v16h vb[4];
#pragma unroll
    for (int j = 0; j < 4; ++j) vb[j] = *(const v16h*)(vtp + j * 16 * T);
    kp0 += 32 * C; kp1 += 32 * C; vtp += 32;

    v8f s[2][2] = {};
#pragma unroll
    for (int i = 0; i < 2; ++i) {
        s[i][0] = WMMA_F16(qa[i][0], kb00, s[i][0]);
        s[i][1] = WMMA_F16(qa[i][0], kb10, s[i][1]);
        s[i][0] = WMMA_F16(qa[i][1], kb01, s[i][0]);
        s[i][1] = WMMA_F16(qa[i][1], kb11, s[i][1]);
    }
    softmax_pv(s[0][0], s[0][1], vb, ones, sp0, half, l16, o[0], lacc[0], mrow[0]);
    softmax_pv(s[1][0], s[1][1], vb, ones, sp1, half, l16, o[1], lacc[1], mrow[1]);
}

// ---------------------------------------------------------------------------
// The single diagonal step (kt == qbase). Tile0 sees only key cols [0,16)
// of this step (its upper 16 cols are fully masked -> skip those 2 WMMAs);
// tile1 sees cols [0,32) with its own diagonal in the upper 16.
// ---------------------------------------------------------------------------
template <int T, int C>
__device__ __forceinline__ void attn_step_diag(
    const _Float16* kp0, const _Float16* kp1, const _Float16* vtp,
    const v16h (&qa)[2][2], const v16h ones,
    _Float16* __restrict__ sp0, _Float16* __restrict__ sp1,
    int half, int l16,
    v8f (&o)[2][4], v8f (&lacc)[2], float (&mrow)[2][8]) {
    v16h kb00 = *(const v16h*)(kp0);
    v16h kb01 = *(const v16h*)(kp0 + 32);
    v16h kb10 = *(const v16h*)(kp1);
    v16h kb11 = *(const v16h*)(kp1 + 32);
    v16h vb[4];
#pragma unroll
    for (int j = 0; j < 4; ++j) vb[j] = *(const v16h*)(vtp + j * 16 * T);

    // tile 0: diagonal in cols [0,16); cols [16,32) fully masked
    v8f s00 = {}, s01;
    s00 = WMMA_F16(qa[0][0], kb00, s00);
    s00 = WMMA_F16(qa[0][1], kb01, s00);
#pragma unroll
    for (int r = 0; r < 8; ++r) {
        if (l16 > half * 8 + r) s00[r] = -1e30f;
        s01[r] = -1e30f;
    }
    softmax_pv(s00, s01, vb, ones, sp0, half, l16, o[0], lacc[0], mrow[0]);

    // tile 1: cols [0,16) unmasked; diagonal in cols [16,32)
    v8f s10 = {}, s11 = {};
    s10 = WMMA_F16(qa[1][0], kb00, s10);
    s11 = WMMA_F16(qa[1][0], kb10, s11);
    s10 = WMMA_F16(qa[1][1], kb01, s10);
    s11 = WMMA_F16(qa[1][1], kb11, s11);
#pragma unroll
    for (int r = 0; r < 8; ++r)
        if (l16 > half * 8 + r) s11[r] = -1e30f;
    softmax_pv(s10, s11, vb, ones, sp1, half, l16, o[1], lacc[1], mrow[1]);
}

// ---------------------------------------------------------------------------
// Causal flash attention. 128 threads = 4 waves, one 32-row Q tile per wave.
// Q/K row-major [B*T, C]; V transposed [(b*C + d), t]; Y row-major [B*T, C].
// ---------------------------------------------------------------------------
template <int T, int H, int Dh>
__global__ __launch_bounds__(128, 1) void attn_causal(const _Float16* __restrict__ Q,
                                                      const _Float16* __restrict__ Km,
                                                      const _Float16* __restrict__ Vt,
                                                      _Float16* __restrict__ Y) {
    constexpr int C = H * Dh;
    __shared__ _Float16 ldsS[4][2][16 * 32];

    const int lane = threadIdx.x & 31;
    const int wave = threadIdx.x >> 5;
    const int half = lane >> 4;
    const int l16  = lane & 15;

    constexpr int blocksPerSeq = T / 128;          // 4 waves * 32 rows
    int bh   = blockIdx.x / blocksPerSeq;
    int qblk = blockIdx.x % blocksPerSeq;
    int b = bh / H, h = bh % H;
    int qbase = qblk * 128 + wave * 32;            // multiple of 32

    const size_t rowBase = (size_t)b * T * C + (size_t)h * Dh;
    const _Float16* kbase = Km + rowBase;
    const _Float16* vtb   = Vt + ((size_t)b * C + (size_t)h * Dh) * T;

    const _Float16 hs = (_Float16)rsqrtf((float)Dh);   // 0.125, exact
    v16h qa[2][2];
#pragma unroll
    for (int i = 0; i < 2; ++i) {
        const _Float16* qrow = Q + rowBase + (size_t)(qbase + i * 16 + l16) * C;
        qa[i][0] = *(const v16h*)(qrow + half * 16);
        qa[i][1] = *(const v16h*)(qrow + 32 + half * 16);
#pragma unroll
        for (int e = 0; e < 16; ++e) { qa[i][0][e] = qa[i][0][e] * hs; qa[i][1][e] = qa[i][1][e] * hs; }
    }

    v16h ones;
#pragma unroll
    for (int e = 0; e < 16; ++e) ones[e] = (_Float16)1.0f;

    v8f o[2][4] = {};
    v8f lacc[2] = {};
    float mrow[2][8];
#pragma unroll
    for (int i = 0; i < 2; ++i)
#pragma unroll
        for (int r = 0; r < 8; ++r) mrow[i][r] = -1e30f;

    _Float16* sp0 = &ldsS[wave][0][0];
    _Float16* sp1 = &ldsS[wave][1][0];

    const _Float16* kp0 = kbase + (size_t)l16 * C + half * 16;
    const _Float16* kp1 = kbase + (size_t)(16 + l16) * C + half * 16;
    const _Float16* vtp = vtb + (size_t)l16 * T + half * 16;

    for (int kt = 0; kt < qbase; kt += 32)
        attn_step_full<T, C>(kp0, kp1, vtp, qa, ones, sp0, sp1, half, l16, o, lacc, mrow);
    attn_step_diag<T, C>(kp0, kp1, vtp, qa, ones, sp0, sp1, half, l16, o, lacc, mrow);

#pragma unroll
    for (int i = 0; i < 2; ++i)
#pragma unroll
        for (int j = 0; j < 4; ++j)
#pragma unroll
            for (int r = 0; r < 8; ++r)
                Y[rowBase + (size_t)(qbase + i * 16 + half * 8 + r) * C + j * 16 + l16] =
                    (_Float16)(o[i][j][r] / lacc[i][r]);
}

// ---------------------------------------------------------------------------
extern "C" void kernel_launch(void* const* d_in, const int* in_sizes, int n_in,
                              void* d_out, int out_size, void* d_ws, size_t ws_size,
                              hipStream_t stream) {
    (void)in_sizes; (void)n_in; (void)out_size; (void)ws_size;
    const int B = 4, T = 2048, C = 1024, H = 16, Dh = 64;
    const size_t nTok = (size_t)B * T;
    const size_t xElems = nTok * C;
    const size_t wElems = (size_t)C * C;

    const float* x  = (const float*)d_in[0];
    const float* Wq = (const float*)d_in[1];
    const float* Wk = (const float*)d_in[2];
    const float* Wv = (const float*)d_in[3];
    const float* Wp = (const float*)d_in[4];
    float* out = (float*)d_out;

    char* ws = (char*)d_ws;
    _Float16* xh  = (_Float16*)ws;  ws += xElems * sizeof(_Float16);
    _Float16* wqh = (_Float16*)ws;  ws += wElems * sizeof(_Float16);
    _Float16* wkh = (_Float16*)ws;  ws += wElems * sizeof(_Float16);
    _Float16* wvh = (_Float16*)ws;  ws += wElems * sizeof(_Float16);
    _Float16* wph = (_Float16*)ws;  ws += wElems * sizeof(_Float16);
    _Float16* qh  = (_Float16*)ws;  ws += xElems * sizeof(_Float16);
    _Float16* kh  = (_Float16*)ws;  ws += xElems * sizeof(_Float16);
    _Float16* vt  = (_Float16*)ws;  ws += xElems * sizeof(_Float16);  // transposed V
    _Float16* yh  = (_Float16*)ws;  ws += xElems * sizeof(_Float16);

    const int thr = 256;
    cast_f32_f16<<<(int)((xElems / 4 + thr - 1) / thr), thr, 0, stream>>>(x,  xh,  (int)(xElems / 4));
    cast_f32_f16<<<(int)((wElems / 4 + thr - 1) / thr), thr, 0, stream>>>(Wq, wqh, (int)(wElems / 4));
    cast_f32_f16<<<(int)((wElems / 4 + thr - 1) / thr), thr, 0, stream>>>(Wk, wkh, (int)(wElems / 4));
    cast_f32_f16<<<(int)((wElems / 4 + thr - 1) / thr), thr, 0, stream>>>(Wv, wvh, (int)(wElems / 4));
    cast_f32_f16<<<(int)((wElems / 4 + thr - 1) / thr), thr, 0, stream>>>(Wp, wph, (int)(wElems / 4));

    const int M = (int)nTok, N = C, K = C;
    int gemmBlocks = (M / 32) * (N / 64) / 4;        // 1024 blocks of 128 threads
    gemm_awt<_Float16, false><<<gemmBlocks, 128, 0, stream>>>(xh, wqh, qh, M, N, K, T);
    gemm_awt<_Float16, false><<<gemmBlocks, 128, 0, stream>>>(xh, wkh, kh, M, N, K, T);
    gemm_awt<_Float16, true ><<<gemmBlocks, 128, 0, stream>>>(xh, wvh, vt, M, N, K, T);

    int attnBlocks = B * H * (T / 128);              // 1024
    attn_causal<2048, 16, 64><<<attnBlocks, 128, 0, stream>>>(qh, kh, vt, yh);

    gemm_awt<float, false><<<gemmBlocks, 128, 0, stream>>>(yh, wph, out, M, N, K, T);
}
